// Multi_GNN_26869315404011
// MI455X (gfx1250) — compile-verified, matrix operand order checked by reading, not run on hardware
//
#include <hip/hip_runtime.h>
#include <hip/hip_bf16.h>
#include <math.h>

#define NN 50000
#define NE 600000
#define NG 512
#define HD 128

typedef __attribute__((ext_vector_type(16))) __bf16 v16bf;
typedef __attribute__((ext_vector_type(8)))  __bf16 v8bf;
typedef __attribute__((ext_vector_type(8)))  float  v8f;

static __device__ __forceinline__ float warp_sum32(float v) {
#pragma unroll
  for (int m = 16; m >= 1; m >>= 1) v += __shfl_xor(v, m, 32);
  return v;
}

static __device__ __forceinline__ unsigned enc_f(float f) {
  unsigned u = __float_as_uint(f);
  return (u & 0x80000000u) ? ~u : (u | 0x80000000u);
}
static __device__ __forceinline__ float dec_f(unsigned u) {
  return (u & 0x80000000u) ? __uint_as_float(u & 0x7FFFFFFFu)
                           : __uint_as_float(~u);
}

// ---------------- small utility kernels ----------------
__global__ void k_fill_f32(float* p, float v, int n) {
  int i = blockIdx.x * blockDim.x + threadIdx.x;
  if (i < n) p[i] = v;
}

__global__ void k_deg_accum(const int* __restrict__ dst, float* deg, int e) {
  int i = blockIdx.x * blockDim.x + threadIdx.x;
  if (i < e) atomicAdd(&deg[dst[i]], 1.0f);
}

__global__ void k_dinv(const float* __restrict__ deg, float* dinv, int n) {
  int i = blockIdx.x * blockDim.x + threadIdx.x;
  if (i < n) dinv[i] = rsqrtf(deg[i]);  // deg >= 1 (self loop)
}

__global__ void k_f2bf(const float* __restrict__ s, __bf16* d, int n) {
  int i = blockIdx.x * blockDim.x + threadIdx.x;
  if (i < n) d[i] = (__bf16)s[i];
}

// Swizzle a 128x128 row-major fp32 weight into WMMA B-fragment order (bf16):
// Wswz[((nt*4+kc)*32+lane)*16 + e] = W[k][n],
//   k = kc*32 + (e/8)*16 + (lane/16)*8 + (e%8),  n = nt*16 + lane%16
__global__ void k_swz_w(const float* __restrict__ W, __bf16* Wswz) {
  int t = blockIdx.x * blockDim.x + threadIdx.x;
  if (t >= 8 * 4 * 32 * 16) return;
  int e = t & 15, lane = (t >> 4) & 31, kc = (t >> 9) & 3, nt = t >> 11;
  int k = kc * 32 + (e >> 3) * 16 + (lane >> 4) * 8 + (e & 7);
  int n = nt * 16 + (lane & 15);
  Wswz[t] = (__bf16)W[k * HD + n];
}

// ---------------- WMMA GEMM: out[rows,128] = A[rows,128](bf16) @ W(128x128 swizzled) ----------------
__global__ void __launch_bounds__(256)
k_gemm_wmma(const __bf16* __restrict__ A, const __bf16* __restrict__ Wswz,
            const float* __restrict__ bias, float* __restrict__ out,
            __bf16* __restrict__ out_bf, int rows, int do_relu) {
  int wave = threadIdx.x >> 5;
  int lane = threadIdx.x & 31;
  int tile = blockIdx.x * 8 + wave;
  if (tile * 16 >= rows) return;
  int m = lane & 15;
  int half = lane >> 4;
  const __bf16* arow = A + (size_t)(tile * 16 + m) * HD;

  v8f acc[8] = {};
#pragma unroll
  for (int kc = 0; kc < 4; ++kc) {
    v8bf alo = *(const v8bf*)(arow + kc * 32 + half * 8);
    v8bf ahi = *(const v8bf*)(arow + kc * 32 + half * 8 + 16);
    v16bf afrag;
#pragma unroll
    for (int e = 0; e < 8; ++e) { afrag[e] = alo[e]; afrag[8 + e] = ahi[e]; }
#pragma unroll
    for (int nt = 0; nt < 8; ++nt) {
      v16bf bfrag = *(const v16bf*)(Wswz + (size_t)((nt * 4 + kc) * 32 + lane) * 16);
      acc[nt] = __builtin_amdgcn_wmma_f32_16x16x32_bf16(
          false, afrag, false, bfrag, (short)0, acc[nt], false, false);
    }
  }
#pragma unroll
  for (int nt = 0; nt < 8; ++nt) {
    int col = nt * 16 + m;
    float bv = bias ? bias[col] : 0.0f;
#pragma unroll
    for (int j = 0; j < 8; ++j) {
      int r = tile * 16 + j + 8 * half;
      float v = acc[nt][j] + bv;
      if (do_relu) v = fmaxf(v, 0.0f);
      out[(size_t)r * HD + col] = v;
      if (out_bf) out_bf[(size_t)r * HD + col] = (__bf16)v;
    }
  }
}

// ---------------- graph aggregation ----------------
// Self-loop term doubles as the aggregation initializer (full overwrite).
__global__ void k_selfloop(const float* __restrict__ h, const float* __restrict__ dinv,
                           float* out, int n) {
  int i = blockIdx.x * blockDim.x + threadIdx.x;
  if (i >= n * HD) return;
  float d = dinv[i >> 7];
  out[i] = h[i] * d * d;
}

__global__ void __launch_bounds__(256)
k_edge_scatter(const int* __restrict__ src, const int* __restrict__ dst,
               const float* __restrict__ dinv, const float* __restrict__ h,
               float* out, int e) {
  int wave = threadIdx.x >> 5, lane = threadIdx.x & 31;
  int edge = blockIdx.x * 8 + wave;
  if (edge >= e) return;
  int s = src[edge], d = dst[edge];
  float nrm = dinv[s] * dinv[d];
  float4 v = ((const float4*)(h + (size_t)s * HD))[lane];
  float* o = out + (size_t)d * HD + lane * 4;
  atomicAdd(o + 0, v.x * nrm);
  atomicAdd(o + 1, v.y * nrm);
  atomicAdd(o + 2, v.z * nrm);
  atomicAdd(o + 3, v.w * nrm);
}

// (agg + bias) -> LayerNorm -> relu ; writes fp32 + bf16 copies. One wave per node.
__global__ void __launch_bounds__(256)
k_bias_ln_relu(const float* __restrict__ agg, const float* __restrict__ bias,
               const float* __restrict__ gam, const float* __restrict__ bet,
               float* out_f, __bf16* out_bf, int n) {
  int wave = threadIdx.x >> 5, lane = threadIdx.x & 31;
  int node = blockIdx.x * 8 + wave;
  if (node >= n) return;
  float4 v  = ((const float4*)(agg + (size_t)node * HD))[lane];
  float4 b4 = ((const float4*)bias)[lane];
  v.x += b4.x; v.y += b4.y; v.z += b4.z; v.w += b4.w;
  float mean = warp_sum32(v.x + v.y + v.z + v.w) * (1.0f / 128.0f);
  float dx = v.x - mean, dy = v.y - mean, dz = v.z - mean, dw = v.w - mean;
  float var = warp_sum32(dx * dx + dy * dy + dz * dz + dw * dw) * (1.0f / 128.0f);
  float inv = rsqrtf(var + 1e-5f);
  float4 g4 = ((const float4*)gam)[lane];
  float4 e4 = ((const float4*)bet)[lane];
  float4 o;
  o.x = fmaxf(dx * inv * g4.x + e4.x, 0.0f);
  o.y = fmaxf(dy * inv * g4.y + e4.y, 0.0f);
  o.z = fmaxf(dz * inv * g4.z + e4.z, 0.0f);
  o.w = fmaxf(dw * inv * g4.w + e4.w, 0.0f);
  ((float4*)(out_f + (size_t)node * HD))[lane] = o;
  __bf16* ob = out_bf + (size_t)node * HD + lane * 4;
  ob[0] = (__bf16)o.x; ob[1] = (__bf16)o.y; ob[2] = (__bf16)o.z; ob[3] = (__bf16)o.w;
}

// gate[n] = dot(row[n], w) + b ; one wave per node
__global__ void __launch_bounds__(256)
k_rowdot(const float* __restrict__ rows, const float* __restrict__ w,
         const float* __restrict__ b, float* out, int n) {
  int wave = threadIdx.x >> 5, lane = threadIdx.x & 31;
  int node = blockIdx.x * 8 + wave;
  if (node >= n) return;
  float4 v = ((const float4*)(rows + (size_t)node * HD))[lane];
  float4 wv = ((const float4*)w)[lane];
  float s = warp_sum32(v.x * wv.x + v.y * wv.y + v.z * wv.z + v.w * wv.w);
  if (lane == 0) out[node] = s + b[0];
}

__global__ void k_init_seg(unsigned* gmax, float* denom, int g) {
  int i = blockIdx.x * blockDim.x + threadIdx.x;
  if (i < g) { gmax[i] = enc_f(-INFINITY); denom[i] = 0.0f; }
}

__global__ void k_segmax(const float* __restrict__ gate, const int* __restrict__ batch,
                         unsigned* gmax, int n) {
  int i = blockIdx.x * blockDim.x + threadIdx.x;
  if (i < n) atomicMax(&gmax[batch[i]], enc_f(gate[i]));
}

__global__ void k_expsum(const float* __restrict__ gate, const int* __restrict__ batch,
                         const unsigned* __restrict__ gmax, float* a, float* denom, int n) {
  int i = blockIdx.x * blockDim.x + threadIdx.x;
  if (i >= n) return;
  int b = batch[i];
  float av = expf(gate[i] - dec_f(gmax[b]));
  a[i] = av;
  atomicAdd(&denom[b], av);
}

__global__ void __launch_bounds__(256)
k_pool(const float* __restrict__ h, const float* __restrict__ a,
       const float* __restrict__ denom, const int* __restrict__ batch,
       float* pooled, int n) {
  int wave = threadIdx.x >> 5, lane = threadIdx.x & 31;
  int node = blockIdx.x * 8 + wave;
  if (node >= n) return;
  int b = batch[node];
  float alpha = a[node] / denom[b];
  float4 v = ((const float4*)(h + (size_t)node * HD))[lane];
  float* o = pooled + (size_t)b * HD + lane * 4;
  atomicAdd(o + 0, v.x * alpha);
  atomicAdd(o + 1, v.y * alpha);
  atomicAdd(o + 2, v.z * alpha);
  atomicAdd(o + 3, v.w * alpha);
}

// logits[g, 0..1] from relu'd classifier hidden; one wave per graph
__global__ void __launch_bounds__(256)
k_cls2(const float* __restrict__ c1, const float* __restrict__ w2,
       const float* __restrict__ b2, float* out, int g) {
  int wave = threadIdx.x >> 5, lane = threadIdx.x & 31;
  int gr = blockIdx.x * 8 + wave;
  if (gr >= g) return;
  float4 v = ((const float4*)(c1 + (size_t)gr * HD))[lane];
  int f = lane * 4;
  float s0 = v.x * w2[(f + 0) * 2] + v.y * w2[(f + 1) * 2] +
             v.z * w2[(f + 2) * 2] + v.w * w2[(f + 3) * 2];
  float s1 = v.x * w2[(f + 0) * 2 + 1] + v.y * w2[(f + 1) * 2 + 1] +
             v.z * w2[(f + 2) * 2 + 1] + v.w * w2[(f + 3) * 2 + 1];
  s0 = warp_sum32(s0);
  s1 = warp_sum32(s1);
  if (lane == 0) { out[gr * 2 + 0] = s0 + b2[0]; out[gr * 2 + 1] = s1 + b2[1]; }
}

// ---------------- host launcher ----------------
extern "C" void kernel_launch(void* const* d_in, const int* in_sizes, int n_in,
                              void* d_out, int out_size, void* d_ws, size_t ws_size,
                              hipStream_t stream) {
  const float* x     = (const float*)d_in[0];
  const int*   eidx  = (const int*)d_in[1];
  const int*   batch = (const int*)d_in[2];
  const float* W1    = (const float*)d_in[3];
  const float* b1    = (const float*)d_in[4];
  const float* ln1g  = (const float*)d_in[5];
  const float* ln1b  = (const float*)d_in[6];
  const float* W2    = (const float*)d_in[7];
  const float* b2    = (const float*)d_in[8];
  const float* ln2g  = (const float*)d_in[9];
  const float* ln2b  = (const float*)d_in[10];
  const float* gw1   = (const float*)d_in[11];
  const float* gb1   = (const float*)d_in[12];
  const float* gw2   = (const float*)d_in[13];
  const float* gb2   = (const float*)d_in[14];
  const float* cw1   = (const float*)d_in[15];
  const float* cb1   = (const float*)d_in[16];
  const float* cw2   = (const float*)d_in[17];
  const float* cb2   = (const float*)d_in[18];
  float* logits = (float*)d_out;

  const int* esrc = eidx;
  const int* edst = eidx + NE;

  size_t off = 0;
  char* base = (char*)d_ws;
  auto alloc = [&](size_t bytes) -> void* {
    void* p = base + off;
    off += (bytes + 255) & ~(size_t)255;
    return p;
  };
  float*    deg      = (float*)alloc((size_t)NN * 4);
  float*    dinv     = (float*)alloc((size_t)NN * 4);
  __bf16*   P0       = (__bf16*)alloc((size_t)NN * HD * 2);  // bf16 activations
  float*    P1       = (float*)alloc((size_t)NN * HD * 4);   // gemm outputs
  float*    P2       = (float*)alloc((size_t)NN * HD * 4);   // aggregation / h_f
  __bf16*   W1s      = (__bf16*)alloc(16384 * 2);
  __bf16*   W2s      = (__bf16*)alloc(16384 * 2);
  __bf16*   GW1s     = (__bf16*)alloc(16384 * 2);
  __bf16*   CW1s     = (__bf16*)alloc(16384 * 2);
  float*    gate     = (float*)alloc((size_t)NN * 4);
  float*    aexp     = (float*)alloc((size_t)NN * 4);
  unsigned* gmax     = (unsigned*)alloc((size_t)NG * 4);
  float*    denom    = (float*)alloc((size_t)NG * 4);
  float*    pooled   = (float*)alloc((size_t)NG * HD * 4);
  __bf16*   pooledbf = (__bf16*)alloc((size_t)NG * HD * 2);
  (void)ws_size; (void)n_in; (void)in_sizes; (void)out_size;

  const int T = 256;
  const int gN    = (NN + T - 1) / T;
  const int gNH   = (NN * HD + T - 1) / T;
  const int gE    = (NE + T - 1) / T;
  const int gSwz  = (8 * 4 * 32 * 16 + T - 1) / T;
  const int gGemN = (NN / 16 + 7) / 8;   // 3125 tiles -> 391 blocks
  const int gGemG = (NG / 16 + 7) / 8;   // 32 tiles -> 4 blocks
  const int gWavN = (NN + 7) / 8;        // wave-per-node kernels
  const int gWavE = (NE + 7) / 8;        // wave-per-edge
  const int gWavG = (NG + 7) / 8;

  // degree / norm
  k_fill_f32<<<gN, T, 0, stream>>>(deg, 1.0f, NN);
  k_deg_accum<<<gE, T, 0, stream>>>(edst, deg, NE);
  k_dinv<<<gN, T, 0, stream>>>(deg, dinv, NN);

  // weight swizzle + input conversion
  k_swz_w<<<gSwz, T, 0, stream>>>(W1, W1s);
  k_swz_w<<<gSwz, T, 0, stream>>>(W2, W2s);
  k_swz_w<<<gSwz, T, 0, stream>>>(gw1, GW1s);
  k_swz_w<<<gSwz, T, 0, stream>>>(cw1, CW1s);
  k_f2bf<<<gNH, T, 0, stream>>>(x, P0, NN * HD);

  // layer 1: GEMM -> scatter -> bias+LN+relu
  k_gemm_wmma<<<gGemN, T, 0, stream>>>(P0, W1s, nullptr, P1, nullptr, NN, 0);
  k_selfloop<<<gNH, T, 0, stream>>>(P1, dinv, P2, NN);
  k_edge_scatter<<<gWavE, T, 0, stream>>>(esrc, edst, dinv, P1, P2, NE);
  k_bias_ln_relu<<<gWavN, T, 0, stream>>>(P2, b1, ln1g, ln1b, P2, P0, NN);

  // layer 2
  k_gemm_wmma<<<gGemN, T, 0, stream>>>(P0, W2s, nullptr, P1, nullptr, NN, 0);
  k_selfloop<<<gNH, T, 0, stream>>>(P1, dinv, P2, NN);
  k_edge_scatter<<<gWavE, T, 0, stream>>>(esrc, edst, dinv, P1, P2, NE);
  k_bias_ln_relu<<<gWavN, T, 0, stream>>>(P2, b2, ln2g, ln2b, P2, P0, NN);
  // P2 = h2 fp32, P0 = h2 bf16

  // attention gate
  k_gemm_wmma<<<gGemN, T, 0, stream>>>(P0, GW1s, gb1, P1, nullptr, NN, 1);
  k_rowdot<<<gWavN, T, 0, stream>>>(P1, gw2, gb2, gate, NN);

  // scatter softmax + weighted pooling
  k_init_seg<<<(NG + T - 1) / T, T, 0, stream>>>(gmax, denom, NG);
  k_segmax<<<gN, T, 0, stream>>>(gate, batch, gmax, NN);
  k_expsum<<<gN, T, 0, stream>>>(gate, batch, gmax, aexp, denom, NN);
  k_fill_f32<<<(NG * HD + T - 1) / T, T, 0, stream>>>(pooled, 0.0f, NG * HD);
  k_pool<<<gWavN, T, 0, stream>>>(P2, aexp, denom, batch, pooled, NN);
  k_f2bf<<<(NG * HD + T - 1) / T, T, 0, stream>>>(pooled, pooledbf, NG * HD);

  // classifier
  k_gemm_wmma<<<gGemG, T, 0, stream>>>(pooledbf, CW1s, cb1, P1, nullptr, NG, 1);
  k_cls2<<<gWavG, T, 0, stream>>>(P1, cw2, cb2, logits, NG);
}
